// UnivNetLvcBlock_64828236365886
// MI455X (gfx1250) — matrix operand order repaired and checked by reading, not compiled, not run on hardware
//
#include <hip/hip_runtime.h>

// ---------------------------------------------------------------------------
// UnivNet LVC block for MI455X (gfx1250, wave32, WMMA).
// Dominant work (77 GF kernel-prediction head) runs as bf16 WMMA GEMM fused
// with the location-variable convolution. Weight rows are pre-permuted so the
// LVC contraction is a lane-local VGPR accumulation (v_fmac co-executing with
// the WMMA pipe), with no LDS atomics in the hot loop.
// ---------------------------------------------------------------------------

typedef __attribute__((ext_vector_type(16))) __bf16 v16bf;
typedef __attribute__((ext_vector_type(8)))  float  v8f;
typedef __attribute__((ext_vector_type(4)))  unsigned int u32x4;
struct Pair128 { u32x4 lo, hi; };   // 32 bytes == v16bf

__device__ __forceinline__ float lrelu(float x) { return x > 0.f ? x : 0.2f * x; }

__device__ __forceinline__ unsigned short f2bf(float f) {
  unsigned int u = __float_as_uint(f);
  return (unsigned short)((u + 0x7FFFu + ((u >> 16) & 1u)) >> 16);   // RNE
}

// ---------------------------------------------------------------------------
// K1: x = convT(lrelu(hidden)), stride 8, k=16, pad 4.  out (8,32,8192)
// y[b,o,t] = b[o] + sum_{i,j} lrelu(h[b,i,j]) * w[o,i,t+4-8j]
// ---------------------------------------------------------------------------
__global__ void convt_kernel(const float* __restrict__ hs, const float* __restrict__ w,
                             const float* __restrict__ bias, float* __restrict__ x)
{
  int gid = blockIdx.x * 256 + threadIdx.x;          // 8*32*8192
  int t  = gid & 8191;
  int co = (gid >> 13) & 31;
  int b  = gid >> 18;
  float acc = bias[co];
  int j1 = (t + 4) >> 3;
#pragma unroll
  for (int dj = 0; dj < 2; dj++) {
    int j = j1 - dj;
    int u = t + 4 - 8 * j;
    if (j >= 0 && j < 1024 && u >= 0 && u < 16) {
      const float* hrow = hs + (size_t)b * 32 * 1024 + j;
      const float* wrow = w + co * 32 * 16 + u;
      for (int ci = 0; ci < 32; ci++)
        acc += lrelu(hrow[ci * 1024]) * wrow[ci * 16];
    }
  }
  x[gid] = acc;
}

// ---------------------------------------------------------------------------
// K2: KP front conv 100->64, k=5, pad 2, lrelu.  out (8,64,1024)
// ---------------------------------------------------------------------------
__global__ void kpin_kernel(const float* __restrict__ spec, const float* __restrict__ w,
                            const float* __restrict__ bias, float* __restrict__ h)
{
  int gid = blockIdx.x * 256 + threadIdx.x;          // 8*64*1024
  int l  = gid & 1023;
  int co = (gid >> 10) & 63;
  int b  = gid >> 16;
  float acc = bias[co];
  const float* sb   = spec + (size_t)b * 100 * 1024;
  const float* wrow = w + co * 100 * 5;
  for (int ci = 0; ci < 100; ci++) {
#pragma unroll
    for (int tau = 0; tau < 5; tau++) {
      int ls = l + tau - 2;
      if (ls >= 0 && ls < 1024)
        acc += sb[ci * 1024 + ls] * wrow[ci * 5 + tau];
    }
  }
  h[gid] = lrelu(acc);
}

// ---------------------------------------------------------------------------
// K3: generic 64->64 k=3 pad 1 conv + lrelu (+ optional residual add)
// ---------------------------------------------------------------------------
__global__ void conv64_kernel(const float* __restrict__ in, float* __restrict__ out,
                              const float* __restrict__ w, const float* __restrict__ bias,
                              const float* __restrict__ res)
{
  int gid = blockIdx.x * 256 + threadIdx.x;          // 8*64*1024
  int l  = gid & 1023;
  int co = (gid >> 10) & 63;
  int b  = gid >> 16;
  float acc = bias[co];
  const float* ib   = in + (size_t)b * 64 * 1024;
  const float* wrow = w + co * 64 * 3;
  for (int ci = 0; ci < 64; ci++) {
#pragma unroll
    for (int tau = 0; tau < 3; tau++) {
      int ls = l + tau - 1;
      if (ls >= 0 && ls < 1024)
        acc += ib[ci * 1024 + ls] * wrow[ci * 3 + tau];
    }
  }
  acc = lrelu(acc);
  if (res) acc += res[gid];
  out[gid] = acc;
}

// ---------------------------------------------------------------------------
// K4: bias head conv 64->256 k=3 pad 1 (no activation). out (8,256,1024)
// ---------------------------------------------------------------------------
__global__ void biashead_kernel(const float* __restrict__ h, float* __restrict__ outb,
                                const float* __restrict__ w, const float* __restrict__ bias)
{
  int gid = blockIdx.x * 256 + threadIdx.x;          // 8*256*1024
  int l  = gid & 1023;
  int ch = (gid >> 10) & 255;
  int b  = gid >> 18;
  float acc = bias[ch];
  const float* ib   = h + (size_t)b * 64 * 1024;
  const float* wrow = w + ch * 64 * 3;
  for (int ci = 0; ci < 64; ci++) {
#pragma unroll
    for (int tau = 0; tau < 3; tau++) {
      int ls = l + tau - 1;
      if (ls >= 0 && ls < 1024)
        acc += ib[ci * 1024 + ls] * wrow[ci * 3 + tau];
    }
  }
  outb[gid] = acc;
}

// ---------------------------------------------------------------------------
// K5: f32 -> bf16 conversion (KP hidden)
// ---------------------------------------------------------------------------
__global__ void cvt_bf16_kernel(const float* __restrict__ in,
                                unsigned short* __restrict__ out, int n)
{
  int gid = blockIdx.x * 256 + threadIdx.x;
  if (gid < n) out[gid] = f2bf(in[gid]);
}

// ---------------------------------------------------------------------------
// K6a: permute + convert predicted-kernel weights.
// New row order (within each dilation block of 6144): r' = ci*192 + kk*64 + o
// (original: (ci*64+o)*3+kk). Columns (K = cin*3+tau = 192) unchanged.
// ---------------------------------------------------------------------------
__global__ void permW_kernel(const float* __restrict__ w, unsigned short* __restrict__ Wb)
{
  int gid = blockIdx.x * 256 + threadIdx.x;          // 24576*192
  int row = gid / 192;
  int k   = gid - row * 192;
  int dil = row / 6144;
  int rr  = row - dil * 6144;
  int ci  = rr / 192;
  int r2  = rr - ci * 192;
  int kk  = r2 >> 6;
  int o   = r2 & 63;
  int orig = ((dil * 32 + ci) * 64 + o) * 3 + kk;
  Wb[gid] = f2bf(w[(size_t)orig * 192 + k]);
}

// K6b: same permutation for the predicted-kernel conv bias (f32)
__global__ void permKb_kernel(const float* __restrict__ kb, float* __restrict__ kbp)
{
  int row = blockIdx.x * 256 + threadIdx.x;          // 24576
  if (row >= 24576) return;
  int dil = row / 6144;
  int rr  = row - dil * 6144;
  int ci  = rr / 192;
  int r2  = rr - ci * 192;
  int kk  = r2 >> 6;
  int o   = r2 & 63;
  kbp[row] = kb[((dil * 32 + ci) * 64 + o) * 3 + kk];
}

// ---------------------------------------------------------------------------
// K7: hpost = lrelu(dilated conv_{d}(lrelu(x)))   (8,32,8192)
// ---------------------------------------------------------------------------
__global__ void dilconv_kernel(const float* __restrict__ x, float* __restrict__ hpost,
                               const float* __restrict__ w, const float* __restrict__ bias,
                               int di, int d)
{
  int gid = blockIdx.x * 256 + threadIdx.x;          // 8*32*8192
  int t = gid & 8191;
  int c = (gid >> 13) & 31;
  int b = gid >> 18;
  float acc = bias[di * 32 + c];
  const float* wrow = w + ((di * 32 + c) * 32) * 3;
  const float* xb   = x + (size_t)b * 32 * 8192;
  for (int ci = 0; ci < 32; ci++) {
#pragma unroll
    for (int tau = 0; tau < 3; tau++) {
      int ts = t + (tau - 1) * d;
      if (ts >= 0 && ts < 8192)
        acc += lrelu(xb[ci * 8192 + ts]) * wrow[ci * 3 + tau];
    }
  }
  hpost[gid] = lrelu(acc);
}

// ---------------------------------------------------------------------------
// K8: fused kernel-prediction GEMM (bf16 WMMA) + LVC + gating + residual.
//
// WG = (b, 16-position l-tile). 8 waves: wave w owns o-block (w&3) and
// ci-half (w>>2) -> 48 chunks/wave, EXEC all-ones at every WMMA.
// Chunk = single (ci,kk), rows o = ob*16..+15 (permuted W layout), so each
// lane's C rows map to its OWN LVC outputs: outp[8][8] accumulates in VGPRs
// (v_fmac co-executes with WMMA). LDS atomics only in the final combine.
// ---------------------------------------------------------------------------
__global__ __launch_bounds__(256) void fused_lvc_kernel(
    const float* __restrict__ xin, float* __restrict__ xout,
    const float* __restrict__ hpost,
    const unsigned short* __restrict__ hb,
    const unsigned short* __restrict__ Wb,
    const float* __restrict__ kbp,
    const float* __restrict__ biasv,
    int dil)
{
  __shared__ float out_acc[64 * 128];                       // 32 KB  (o, t-in-tile)
  __shared__ __align__(16) float hin[32 * 132];             // 16.9 KB (ci, window)
  __shared__ __align__(16) unsigned short Bpan[16 * 200];   // 6.4 KB (col stride 400B)

  const int b     = blockIdx.x >> 6;     // 8 batches
  const int tile  = blockIdx.x & 63;     // 64 tiles of 16 l's
  const int lbase = tile * 16;
  const int tid   = threadIdx.x;

  for (int idx = tid; idx < 64 * 128; idx += 256) out_acc[idx] = 0.f;
  for (int idx = tid; idx < 32 * 132; idx += 256) {
    int ci = idx / 132, p = idx - ci * 132;
    int t = lbase * 8 - 1 + p;                             // LVC pad = 1
    hin[idx] = (p < 130 && t >= 0 && t < 8192)
             ? hpost[(b * 32 + ci) * 8192 + t] : 0.f;
  }
  for (int idx = tid; idx < 16 * 192; idx += 256) {        // im2col B panel
    int n = idx / 192, k = idx - n * 192;
    int cin = k / 3, tau = k - cin * 3;
    int ls = lbase + n + tau - 1;
    unsigned short v = 0;
    if (ls >= 0 && ls < 1024) v = hb[(b * 64 + cin) * 1024 + ls];
    Bpan[n * 200 + k] = v;
  }
  __syncthreads();

  const int lane = tid & 31;
  const int wave = tid >> 5;
  const int lcol = lane & 15;            // N column (l within tile)
  const int hsel = lane >> 4;            // lane half (K split per ISA layout)
  const int ob   = wave & 3;             // o-block: o = ob*16 + vi + 8*hsel
  const int cih  = wave >> 2;            // ci half: ci = cih*16 + ci2

  // B operand (K=192 -> 6 steps of 32), VGPR-resident for the whole loop.
  v16bf Breg[6];
  {
    const u32x4* colp = (const u32x4*)&Bpan[lcol * 200];
#pragma unroll
    for (int ks = 0; ks < 6; ks++) {
      Pair128 q;
      q.lo = colp[ks * 4 + hsel];
      q.hi = colp[ks * 4 + hsel + 2];
      Breg[ks] = __builtin_bit_cast(v16bf, q);
    }
  }

  float outp[8][8];                      // lane-private LVC accumulator
#pragma unroll
  for (int i = 0; i < 8; i++)
#pragma unroll
    for (int j = 0; j < 8; j++) outp[i][j] = 0.f;

  const unsigned short* Wd  = Wb  + (size_t)dil * 6144 * 192;
  const float*          kbd = kbp + dil * 6144;

  for (int ci2 = 0; ci2 < 16; ci2++) {
    const int ci = cih * 16 + ci2;
    // LVC input values this lane needs for all 3 kk-chunks of this ci
    const float4 h0 = *(const float4*)&hin[ci * 132 + lcol * 8 + 0];
    const float4 h1 = *(const float4*)&hin[ci * 132 + lcol * 8 + 4];
    const float4 h2 = *(const float4*)&hin[ci * 132 + lcol * 8 + 8];
    const float hv[12] = { h0.x, h0.y, h0.z, h0.w, h1.x, h1.y, h1.z, h1.w,
                           h2.x, h2.y, h2.z, h2.w };
#pragma unroll
    for (int kk = 0; kk < 3; kk++) {
      const int rbase = ci * 192 + kk * 64 + ob * 16;      // chunk-uniform (SALU)
      const u32x4* wq = (const u32x4*)(Wd + (size_t)(rbase + lcol) * 192);
      v8f acc = {0.f, 0.f, 0.f, 0.f, 0.f, 0.f, 0.f, 0.f};
#pragma unroll
      for (int ks = 0; ks < 6; ks++) {
        Pair128 qa;                                        // 16-bit A layout:
        qa.lo = wq[ks * 4 + hsel];                         //  K0-7  / K8-15
        qa.hi = wq[ks * 4 + hsel + 2];                     //  K16-23/ K24-31
        v16bf Areg = __builtin_bit_cast(v16bf, qa);
        acc = __builtin_amdgcn_wmma_f32_16x16x32_bf16(false, Areg, false, Breg[ks],
                                                      (short)0, acc, false, false);
      }
      const float4 kb0 = *(const float4*)&kbd[rbase + hsel * 8 + 0];
      const float4 kb1 = *(const float4*)&kbd[rbase + hsel * 8 + 4];
      const float kbv[8] = { kb0.x, kb0.y, kb0.z, kb0.w,
                             kb1.x, kb1.y, kb1.z, kb1.w };
      // Lane-local LVC accumulation: C row vi+8*hsel -> o = ob*16+vi+8*hsel,
      // column lcol -> t = lcol*8+s. Pure VALU, no LDS.
#pragma unroll
      for (int vi = 0; vi < 8; vi++) {
        const float val = acc[vi] + kbv[vi];               // + predicted bias
#pragma unroll
        for (int s = 0; s < 8; s++)
          outp[vi][s] = __builtin_fmaf(val, hv[kk + s], outp[vi][s]);
      }
    }
  }

  // One-time combine: 2 ci-half waves per o-block -> ds_add_f32
#pragma unroll
  for (int vi = 0; vi < 8; vi++) {
    const int o = ob * 16 + hsel * 8 + vi;
#pragma unroll
    for (int s = 0; s < 8; s++)
      atomicAdd(&out_acc[o * 128 + lcol * 8 + s], outp[vi][s]);
  }
  __syncthreads();

  // bias + gated activation + residual
  for (int idx = tid; idx < 32 * 128; idx += 256) {
    int co = idx >> 7, tt = idx & 127;
    int l = lbase + (tt >> 3);
    int t = lbase * 8 + tt;
    float za = out_acc[co * 128 + tt]
             + biasv[(size_t)(b * 256 + dil * 64 + co) * 1024 + l];
    float zb = out_acc[(co + 32) * 128 + tt]
             + biasv[(size_t)(b * 256 + dil * 64 + co + 32) * 1024 + l];
    float g = (1.f / (1.f + __expf(-za))) * tanhf(zb);
    size_t oi = (size_t)(b * 32 + co) * 8192 + t;
    xout[oi] = xin[oi] + g;
  }
}

// ---------------------------------------------------------------------------
extern "C" void kernel_launch(void* const* d_in, const int* in_sizes, int n_in,
                              void* d_out, int out_size, void* d_ws, size_t ws_size,
                              hipStream_t stream)
{
  (void)in_sizes; (void)n_in; (void)out_size; (void)ws_size;

  const float* hs          = (const float*)d_in[0];
  const float* spec        = (const float*)d_in[1];
  const float* convt_w     = (const float*)d_in[2];
  const float* convt_b     = (const float*)d_in[3];
  const float* kp_in_w     = (const float*)d_in[4];
  const float* kp_in_b     = (const float*)d_in[5];
  const float* kp_res_w1   = (const float*)d_in[6];
  const float* kp_res_b1   = (const float*)d_in[7];
  const float* kp_res_w2   = (const float*)d_in[8];
  const float* kp_res_b2   = (const float*)d_in[9];
  const float* kp_kernel_w = (const float*)d_in[10];
  const float* kp_kernel_b = (const float*)d_in[11];
  const float* kp_bias_w   = (const float*)d_in[12];
  const float* kp_bias_b   = (const float*)d_in[13];
  const float* lvc_w       = (const float*)d_in[14];
  const float* lvc_b       = (const float*)d_in[15];

  char* ws = (char*)d_ws;
  float*          xbuf  = (float*)(ws);                          // 8 MB (8,32,8192)
  float*          hpost = (float*)(ws + (size_t)( 8u << 20));    // 8 MB (8,32,8192)
  float*          bufH  = (float*)(ws + (size_t)(16u << 20));    // 2 MB (8,64,1024)
  float*          bufT  = (float*)(ws + (size_t)(18u << 20));    // 2 MB
  float*          biasv = (float*)(ws + (size_t)(20u << 20));    // 8 MB (8,256,1024)
  unsigned short* hb    = (unsigned short*)(ws + (size_t)(28u << 20)); // 1 MB bf16
  unsigned short* Wb    = (unsigned short*)(ws + (size_t)(29u << 20)); // 9 MB bf16 (permuted)
  float*          kbp   = (float*)(ws + (size_t)(38u << 20));    // 96 KB f32 (permuted)

  // Stage 1: pre-net convtranspose
  convt_kernel<<<8192, 256, 0, stream>>>(hs, convt_w, convt_b, xbuf);

  // Stage 2: kernel-predictor trunk
  kpin_kernel<<<2048, 256, 0, stream>>>(spec, kp_in_w, kp_in_b, bufH);
  for (int r = 0; r < 3; r++) {
    conv64_kernel<<<2048, 256, 0, stream>>>(bufH, bufT, kp_res_w1 + r * 64 * 64 * 3,
                                            kp_res_b1 + r * 64, nullptr);
    conv64_kernel<<<2048, 256, 0, stream>>>(bufT, bufH, kp_res_w2 + r * 64 * 64 * 3,
                                            kp_res_b2 + r * 64, bufH);
  }
  biashead_kernel<<<8192, 256, 0, stream>>>(bufH, biasv, kp_bias_w, kp_bias_b);

  // Stage 3: bf16 / permuted operands for the WMMA GEMM
  cvt_bf16_kernel<<<2048,  256, 0, stream>>>(bufH, hb, 8 * 64 * 1024);
  permW_kernel<<<18432, 256, 0, stream>>>(kp_kernel_w, Wb);
  permKb_kernel<<<96, 256, 0, stream>>>(kp_kernel_b, kbp);

  // Stage 4: four sequential LVC residual blocks
  const int dils[4] = {1, 3, 9, 27};
  for (int i = 0; i < 4; i++) {
    dilconv_kernel<<<8192, 256, 0, stream>>>(xbuf, hpost, lvc_w, lvc_b, i, dils[i]);
    float* xo = (i == 3) ? (float*)d_out : xbuf;
    fused_lvc_kernel<<<512, 256, 0, stream>>>(xbuf, xo, hpost, hb, Wb,
                                              kbp, biasv, i);
  }
}
